// GPT2_70119636075211
// MI455X (gfx1250) — matrix operand order repaired and checked by reading, not compiled, not run on hardware
//
#include <hip/hip_runtime.h>
#include <hip/hip_bf16.h>
#include <math.h>

// ---------------- problem constants (GPT-2 small fwd) ----------------
#define LAYERS 12
#define BB 2
#define TT 1024
#define CC 768
#define HH 12
#define DD 64
#define VV 50257
#define MTOK (BB * TT)          // 2048 tokens
#define EPSLN 1e-3f

// ---------------- CDNA5 WMMA types ----------------
typedef __attribute__((ext_vector_type(16))) __bf16 v16bf;
typedef __attribute__((ext_vector_type(8)))  __bf16 v8bf;
typedef __attribute__((ext_vector_type(8)))  float  v8f;

__device__ inline v8f wmma_bf16(v16bf a, v16bf b, v8f c) {
  // 8 args: (neg_a, A, neg_b, B, c_mod, C, reuse_a, reuse_b)
  return __builtin_amdgcn_wmma_f32_16x16x32_bf16(false, a, false, b, (short)0, c, false, false);
}

// A-fragment (16x32 bf16, M x K) per CDNA5 ISA layout:
// lane = {r = lane&15 (row M), hi = lane>>4}; lane holds K in [8*hi,8*hi+8) and [16+8*hi,16+8*hi+8)
__device__ inline v16bf make_afrag(const __bf16* base, int stride, int lane) {
  int r = lane & 15, hi = lane >> 4;
  const __bf16* row = base + r * stride;
  v8bf lo = *(const v8bf*)(row + 8 * hi);
  v8bf up = *(const v8bf*)(row + 16 + 8 * hi);
  v16bf a;
#pragma unroll
  for (int i = 0; i < 8; ++i) { a[i] = lo[i]; a[8 + i] = up[i]; }
  return a;
}

// B-fragment (32x16 bf16, K x N) from an N-major LDS tile:
// lane = {n = lane&15 (col N), hi = lane>>4}; lane holds K in [8*hi,8*hi+8) and [16+8*hi,16+8*hi+8)
__device__ inline v16bf make_bfrag(const __bf16* baseNmajor, int stride, int lane) {
  int n = lane & 15, hi = lane >> 4;
  const __bf16* row = baseNmajor + n * stride + 8 * hi;
  v8bf lo = *(const v8bf*)(row);
  v8bf up = *(const v8bf*)(row + 16);
  v16bf b;
#pragma unroll
  for (int i = 0; i < 8; ++i) { b[i] = lo[i]; b[8 + i] = up[i]; }
  return b;
}

// ---------------- conversion kernels ----------------
__global__ void f32_to_bf16_kernel(const float* __restrict__ in, __bf16* __restrict__ out,
                                   long long n) {
  long long i = ((long long)blockIdx.x * blockDim.x + threadIdx.x) * 4;
  if (i + 3 < n) {
    float4 v = *(const float4*)(in + i);
    out[i + 0] = (__bf16)v.x; out[i + 1] = (__bf16)v.y;
    out[i + 2] = (__bf16)v.z; out[i + 3] = (__bf16)v.w;
  } else {
    for (; i < n; ++i) out[i] = (__bf16)in[i];
  }
}

// fp32 weight [l][K][N] -> bf16 [l][N][K] (N-major for GEMM B staging).
// 32x32 tiles via LDS, both directions coalesced. K,N multiples of 32.
__global__ __launch_bounds__(256)
void convert_transpose_kernel(const float* __restrict__ in, __bf16* __restrict__ out,
                              int K, int N) {
  __shared__ float tile[32][33];
  int l = blockIdx.z;
  const float* inL = in + (size_t)l * K * N;
  __bf16* outL = out + (size_t)l * K * N;
  int n0 = blockIdx.x * 32, k0 = blockIdx.y * 32;
  int tx = threadIdx.x & 31, ty = threadIdx.x >> 5;   // ty = 0..7
#pragma unroll
  for (int i = 0; i < 4; ++i)
    tile[ty + 8 * i][tx] = inL[(size_t)(k0 + ty + 8 * i) * N + n0 + tx];
  __syncthreads();
#pragma unroll
  for (int i = 0; i < 4; ++i)
    outL[(size_t)(n0 + ty + 8 * i) * K + k0 + tx] = (__bf16)tile[tx][ty + 8 * i];
}

__global__ void embed_kernel(const int* __restrict__ ids, const float* __restrict__ wte,
                             const float* __restrict__ wpe, float* __restrict__ x) {
  int i = blockIdx.x * blockDim.x + threadIdx.x;
  if (i >= MTOK * CC) return;
  int tok = i / CC, c = i - tok * CC;
  int t = tok & (TT - 1);
  int id = ids[tok];
  x[i] = wte[(size_t)id * CC + c] + wpe[(size_t)t * CC + c];
}

// LayerNorm over C=768, one row per block (256 threads, 3 elems each), bf16 out.
__global__ __launch_bounds__(256)
void layernorm_kernel(const float* __restrict__ x, const float* __restrict__ g,
                      const float* __restrict__ b, __bf16* __restrict__ out) {
  __shared__ float s1[256];
  __shared__ float s2[256];
  int row = blockIdx.x, tid = threadIdx.x;
  const float* xr = x + (size_t)row * CC;
  float v0 = xr[tid], v1 = xr[tid + 256], v2 = xr[tid + 512];
  s1[tid] = v0 + v1 + v2;
  s2[tid] = v0 * v0 + v1 * v1 + v2 * v2;
  __syncthreads();
  for (int off = 128; off > 0; off >>= 1) {
    if (tid < off) { s1[tid] += s1[tid + off]; s2[tid] += s2[tid + off]; }
    __syncthreads();
  }
  float mu = s1[0] * (1.0f / CC);
  float var = s2[0] * (1.0f / CC) - mu * mu;     // jnp.var semantics
  float rstd = rsqrtf(var + EPSLN);
  __bf16* orow = out + (size_t)row * CC;
#pragma unroll
  for (int i = 0; i < 3; ++i) {
    int c = tid + i * 256;
    float xv = (i == 0) ? v0 : (i == 1) ? v1 : v2;
    orow[c] = (__bf16)((xv - mu) * rstd * g[c] + b[c]);
  }
}

// ---------------- tiled bf16 WMMA GEMM (double-buffered) ----------------
// C[M,N] = act(A[M,K] @ B + bias) (+Res); A row-major bf16, B stored N-major [N][K].
// Block 256 thr = 8 waves; tile 128(M) x 128(N); wave owns 32x64 (8 wmma / K-step).

__device__ inline void stageA(__bf16* As, const __bf16* __restrict__ A,
                              int m0, int k0, int K, int tid) {
  int row = tid >> 1, half = tid & 1;
  const __bf16* g = A + (size_t)(m0 + row) * K + k0 + half * 16;
  *(v8bf*)&As[row * 40 + half * 16]     = *(const v8bf*)g;
  *(v8bf*)&As[row * 40 + half * 16 + 8] = *(const v8bf*)(g + 8);
}

__device__ inline void stageB(__bf16* Bt, const __bf16* __restrict__ Bm,
                              int n0, int k0, int N, int K, int tid) {
  int n = tid >> 1, koff = (tid & 1) * 16;
  v8bf v0 = {}, v1 = {};
  if (n0 + n < N) {                // tail guard only bites on logits V-edge
    const __bf16* g = Bm + (size_t)(n0 + n) * K + k0 + koff;
    v0 = *(const v8bf*)g; v1 = *(const v8bf*)(g + 8);
  }
  *(v8bf*)&Bt[n * 40 + koff]     = v0;
  *(v8bf*)&Bt[n * 40 + koff + 8] = v1;
}

template <bool GELU_, bool RES, bool OUTF, bool OUTB>
__global__ __launch_bounds__(256)
void gemm_kernel(const __bf16* __restrict__ A, const __bf16* __restrict__ Bm,
                 const float* __restrict__ bias, const float* __restrict__ Res,
                 float* __restrict__ outF, __bf16* __restrict__ outB,
                 int M, int N, int K) {
  __attribute__((aligned(16))) __shared__ __bf16 As[2][128 * 40];
  __attribute__((aligned(16))) __shared__ __bf16 Bt[2][128 * 40];   // N-major

  int m0 = blockIdx.x * 128;
  int n0 = blockIdx.y * 128;
  int tid = threadIdx.x;
  int wave = tid >> 5, lane = tid & 31;
  int wm = (wave & 3) * 32;                       // wave M offset in tile
  int wn = (wave >> 2) * 64;                      // wave N offset in tile

  v8f acc[2][4] = {};
  int nk = K >> 5;

  stageA(As[0], A, m0, 0, K, tid);
  stageB(Bt[0], Bm, n0, 0, N, K, tid);
  __syncthreads();

  for (int ki = 0; ki < nk; ++ki) {
    int cur = ki & 1;
    if (ki + 1 < nk) {
      stageA(As[cur ^ 1], A, m0, (ki + 1) * 32, K, tid);
      stageB(Bt[cur ^ 1], Bm, n0, (ki + 1) * 32, N, K, tid);
      if (ki + 2 < nk) {  // prefetch K+2 tiles into cache (global_prefetch_b8)
        __builtin_prefetch(A + (size_t)(m0 + (tid >> 1)) * K + (ki + 2) * 32, 0, 1);
        if (n0 + (tid >> 1) < N)
          __builtin_prefetch(Bm + (size_t)(n0 + (tid >> 1)) * K + (ki + 2) * 32, 0, 1);
      }
    }
    const __bf16* Ac = As[cur];
    const __bf16* Bc = Bt[cur];
    v16bf af[2], bf_[4];
    af[0] = make_afrag(Ac + wm * 40, 40, lane);
    af[1] = make_afrag(Ac + (wm + 16) * 40, 40, lane);
#pragma unroll
    for (int j = 0; j < 4; ++j) bf_[j] = make_bfrag(Bc + (wn + j * 16) * 40, 40, lane);
#pragma unroll
    for (int fm = 0; fm < 2; ++fm)
#pragma unroll
      for (int fn = 0; fn < 4; ++fn)
        acc[fm][fn] = wmma_bf16(af[fm], bf_[fn], acc[fm][fn]);
    __syncthreads();
  }

  // epilogue: C/D layout M = e + 8*hi, N = lane&15
  int cn = lane & 15, hi = lane >> 4;
#pragma unroll
  for (int fm = 0; fm < 2; ++fm) {
#pragma unroll
    for (int fn = 0; fn < 4; ++fn) {
      int gm = m0 + wm + fm * 16 + hi * 8;
      int gn = n0 + wn + fn * 16 + cn;
      if (gn < N) {
        float bv = bias ? bias[gn] : 0.f;
#pragma unroll
        for (int e = 0; e < 8; ++e) {
          size_t idx = (size_t)(gm + e) * N + gn;
          float v = acc[fm][fn][e] + bv;
          if (GELU_) v = 0.5f * v * (1.0f + erff(v * 0.70710678118654752f));
          if (RES) v += Res[idx];
          if (OUTF) outF[idx] = v;
          if (OUTB) outB[idx] = (__bf16)v;
        }
      }
    }
  }
}

// ---------------- flash attention (causal), WMMA ----------------
// One block = 128 thr (4 waves) handles one (b, h, 16-query tile).
// Scores and P*V both via v_wmma_f32_16x16x32_bf16; online softmax in LDS.
__global__ __launch_bounds__(128)
void attention_kernel(const __bf16* __restrict__ qkv, __bf16* __restrict__ outB) {
  const int SQ = 72, SC = 72, SP = 72, SV = 72;
  __attribute__((aligned(16))) __shared__ __bf16 Qs[16 * 72];
  __attribute__((aligned(16))) __shared__ float  Sc[16 * 72];   // score chunk fp32 (16x64)
  __attribute__((aligned(16))) __shared__ __bf16 Pc[16 * 72];   // prob chunk bf16
  __attribute__((aligned(16))) __shared__ __bf16 Vt[64 * 72];   // V^T chunk: [channel][key]
  __shared__ float red[2 * 16 * 8];
  __shared__ float mrow[16], lrow[16], fac[16];

  int tid = threadIdx.x, wave = tid >> 5, lane = tid & 31;
  int qb = blockIdx.x & 63;
  int h  = (blockIdx.x >> 6) % HH;
  int b  = blockIdx.x / (64 * HH);
  const size_t base = (size_t)b * TT;
  const int LD = 3 * CC;
  const float scale = 0.125f;                    // 1/sqrt(64)
  int n = lane & 15, hi = lane >> 4;

  { // stage Q tile (16 x 64)
    int qr = tid >> 3, c0 = (tid & 7) * 8;
    *(v8bf*)&Qs[qr * SQ + c0] =
        *(const v8bf*)(qkv + (base + qb * 16 + qr) * LD + h * DD + c0);
  }
  if (tid < 16) { mrow[tid] = -1e30f; lrow[tid] = 0.f; }

  v16bf aq0, aq1;
  v8f acco = {};
  int kc_count = (16 * (qb + 1) + 63) >> 6;      // 64-key chunks (causal bound)

  for (int kc = 0; kc < kc_count; ++kc) {
    __syncthreads();                              // also covers Qs/mrow init on kc==0
    if (kc == 0) { aq0 = make_afrag(Qs, SQ, lane); aq1 = make_afrag(Qs + 32, SQ, lane); }

    { // stage V^T chunk: Vt[c][k] = V[token kc*64+k][c]
      int k = tid >> 1, c0 = (tid & 1) * 32;
      const __bf16* vr = qkv + (base + kc * 64 + k) * LD + 2 * CC + h * DD + c0;
#pragma unroll
      for (int j = 0; j < 4; ++j) {
        v8bf v = *(const v8bf*)(vr + j * 8);
#pragma unroll
        for (int i = 0; i < 8; ++i) Vt[(c0 + j * 8 + i) * SV + k] = v[i];
      }
    }
    { // scores: wave computes 16 queries x 16 keys; K-dim = 64 channels (2 wmma)
      int key = kc * 64 + wave * 16 + n;
      const __bf16* krow = qkv + (base + key) * LD + CC + h * DD;
      v16bf bk0, bk1;
      {
        v8bf l0 = *(const v8bf*)(krow + 8 * hi);
        v8bf l1 = *(const v8bf*)(krow + 16 + 8 * hi);
        v8bf l2 = *(const v8bf*)(krow + 32 + 8 * hi);
        v8bf l3 = *(const v8bf*)(krow + 48 + 8 * hi);
#pragma unroll
        for (int i = 0; i < 8; ++i) {
          bk0[i] = l0[i]; bk0[8 + i] = l1[i];
          bk1[i] = l2[i]; bk1[8 + i] = l3[i];
        }
      }
      v8f sc = {};
      sc = wmma_bf16(aq0, bk0, sc);
      sc = wmma_bf16(aq1, bk1, sc);
#pragma unroll
      for (int e = 0; e < 8; ++e) {
        int m = 8 * hi + e;
        float s = sc[e] * scale;
        if (key > qb * 16 + m) s = -1e30f;        // causal mask
        Sc[m * SC + wave * 16 + n] = s;
      }
    }
    __syncthreads();

    // online softmax over this 64-key chunk: 8 threads per row
    int row = tid >> 3, seg = tid & 7;
    {
      float cm = -1e30f;
#pragma unroll
      for (int j = seg; j < 64; j += 8) cm = fmaxf(cm, Sc[row * SC + j]);
      red[row * 8 + seg] = cm;
    }
    __syncthreads();
    if (seg == 0) {
      float cm = red[row * 8];
#pragma unroll
      for (int i = 1; i < 8; ++i) cm = fmaxf(cm, red[row * 8 + i]);
      float nm = fmaxf(mrow[row], cm);
      fac[row] = expf(mrow[row] - nm);
      mrow[row] = nm;
    }
    __syncthreads();
    {
      float nm = mrow[row], ps = 0.f;
#pragma unroll
      for (int j = seg; j < 64; j += 8) {
        float pv = expf(Sc[row * SC + j] - nm);   // masked (-1e30) -> 0
        Pc[row * SP + j] = (__bf16)pv;
        ps += pv;
      }
      red[128 + row * 8 + seg] = ps;
    }
    __syncthreads();
    if (seg == 0) {
      float s = 0.f;
#pragma unroll
      for (int i = 0; i < 8; ++i) s += red[128 + row * 8 + i];
      lrow[row] = lrow[row] * fac[row] + s;
    }
    __syncthreads();

    // rescale running output and accumulate P @ V (wave owns 16 channels)
#pragma unroll
    for (int e = 0; e < 8; ++e) acco[e] *= fac[8 * hi + e];
    v16bf ap0 = make_afrag(Pc, SP, lane);
    v16bf ap1 = make_afrag(Pc + 32, SP, lane);
    v16bf bv0 = make_bfrag(&Vt[(wave * 16) * SV], SV, lane);
    v16bf bv1 = make_bfrag(&Vt[(wave * 16) * SV + 32], SV, lane);
    acco = wmma_bf16(ap0, bv0, acco);
    acco = wmma_bf16(ap1, bv1, acco);
  }

  // normalize and write attention output (bf16)
#pragma unroll
  for (int e = 0; e < 8; ++e) {
    int m = 8 * hi + e;
    float v = acco[e] / lrow[m];
    outB[(base + qb * 16 + m) * CC + h * DD + wave * 16 + n] = (__bf16)v;
  }
}

// ---------------- host-side orchestration ----------------
extern "C" void kernel_launch(void* const* d_in, const int* in_sizes, int n_in,
                              void* d_out, int out_size, void* d_ws, size_t ws_size,
                              hipStream_t stream) {
  (void)in_sizes; (void)n_in; (void)out_size; (void)ws_size;
  const int*   ids    = (const int*)d_in[0];
  const float* wte    = (const float*)d_in[1];
  const float* wpe    = (const float*)d_in[2];
  const float* ln1_g  = (const float*)d_in[3];
  const float* ln1_b  = (const float*)d_in[4];
  const float* attn_w = (const float*)d_in[5];
  const float* attn_b = (const float*)d_in[6];
  const float* apw    = (const float*)d_in[7];
  const float* apb    = (const float*)d_in[8];
  const float* ln2_g  = (const float*)d_in[9];
  const float* ln2_b  = (const float*)d_in[10];
  const float* fcw    = (const float*)d_in[11];
  const float* fcb    = (const float*)d_in[12];
  const float* pw     = (const float*)d_in[13];
  const float* pb     = (const float*)d_in[14];
  const float* lnf_g  = (const float*)d_in[15];
  const float* lnf_b  = (const float*)d_in[16];

  // carve workspace (~285 MB); all weight buffers hold bf16 in N-major [N][K] layout
  char* p = (char*)d_ws;
  auto alloc = [&](size_t bytes) { char* r = p; p += (bytes + 255) & ~(size_t)255; return r; };
  __bf16* attnw_bf = (__bf16*)alloc((size_t)LAYERS * CC * 3 * CC * 2);
  __bf16* apw_bf   = (__bf16*)alloc((size_t)LAYERS * CC * CC * 2);
  __bf16* fcw_bf   = (__bf16*)alloc((size_t)LAYERS * CC * 4 * CC * 2);
  __bf16* pw_bf    = (__bf16*)alloc((size_t)LAYERS * 4 * CC * CC * 2);
  __bf16* wte_bf   = (__bf16*)alloc((size_t)VV * CC * 2);      // [V][C] already N-major
  float*  xbuf     = (float*) alloc((size_t)MTOK * CC * 4);
  __bf16* hbuf     = (__bf16*)alloc((size_t)MTOK * CC * 2);
  __bf16* qkv_bf   = (__bf16*)alloc((size_t)MTOK * 3 * CC * 2);
  __bf16* att_bf   = (__bf16*)alloc((size_t)MTOK * CC * 2);
  __bf16* mlp_bf   = (__bf16*)alloc((size_t)MTOK * 4 * CC * 2);

  // weight conversion: fp32 [l][K][N] -> bf16 [l][N][K]
  auto cvtT = [&](const float* in, __bf16* out, int K, int N) {
    convert_transpose_kernel<<<dim3(N / 32, K / 32, LAYERS), 256, 0, stream>>>(in, out, K, N);
  };
  cvtT(attn_w, attnw_bf, CC, 3 * CC);
  cvtT(apw,    apw_bf,   CC, CC);
  cvtT(fcw,    fcw_bf,   CC, 4 * CC);
  cvtT(pw,     pw_bf,    4 * CC, CC);
  { // wte: plain convert (already [V][C])
    long long nElem = (long long)VV * CC;
    f32_to_bf16_kernel<<<dim3((unsigned)((nElem + 1023) / 1024)), 256, 0, stream>>>(
        wte, wte_bf, nElem);
  }

  embed_kernel<<<(MTOK * CC + 255) / 256, 256, 0, stream>>>(ids, wte, wpe, xbuf);

  dim3 blk(256);
  for (int l = 0; l < LAYERS; ++l) {
    layernorm_kernel<<<MTOK, blk, 0, stream>>>(xbuf, ln1_g + l * CC, ln1_b + l * CC, hbuf);
    // qkv = h @ Wqkv + b  -> bf16
    gemm_kernel<false, false, false, true>
        <<<dim3(MTOK / 128, (3 * CC) / 128), blk, 0, stream>>>(
            hbuf, attnw_bf + (size_t)l * CC * 3 * CC, attn_b + (size_t)l * 3 * CC,
            nullptr, nullptr, qkv_bf, MTOK, 3 * CC, CC);
    attention_kernel<<<BB * HH * (TT / 16), 128, 0, stream>>>(qkv_bf, att_bf);
    // x += att @ Wproj + b
    gemm_kernel<false, true, true, false>
        <<<dim3(MTOK / 128, CC / 128), blk, 0, stream>>>(
            att_bf, apw_bf + (size_t)l * CC * CC, apb + (size_t)l * CC,
            xbuf, xbuf, nullptr, MTOK, CC, CC);
    layernorm_kernel<<<MTOK, blk, 0, stream>>>(xbuf, ln2_g + l * CC, ln2_b + l * CC, hbuf);
    // m = gelu(h2 @ Wfc + b) -> bf16
    gemm_kernel<true, false, false, true>
        <<<dim3(MTOK / 128, (4 * CC) / 128), blk, 0, stream>>>(
            hbuf, fcw_bf + (size_t)l * CC * 4 * CC, fcb + (size_t)l * 4 * CC,
            nullptr, nullptr, mlp_bf, MTOK, 4 * CC, CC);
    // x += m @ Wproj2 + b
    gemm_kernel<false, true, true, false>
        <<<dim3(MTOK / 128, CC / 128), blk, 0, stream>>>(
            mlp_bf, pw_bf + (size_t)l * 4 * CC * CC, pb + (size_t)l * CC,
            xbuf, xbuf, nullptr, MTOK, CC, 4 * CC);
  }

  layernorm_kernel<<<MTOK, blk, 0, stream>>>(xbuf, lnf_g, lnf_b, hbuf);
  // logits = xf @ wte^T (wte is [V][C] N-major), no bias, fp32 out
  gemm_kernel<false, false, true, false>
      <<<dim3(MTOK / 128, (VV + 127) / 128), blk, 0, stream>>>(
          hbuf, wte_bf, nullptr, nullptr, (float*)d_out, nullptr, MTOK, VV, CC);
}
